// HAN_48541720379811
// MI455X (gfx1250) — matrix operand order repaired and checked by reading, not compiled, not run on hardware
//
#include <hip/hip_runtime.h>

// ---------------------------------------------------------------------------
// Types for CDNA5 WMMA (wave32, v_wmma_f32_16x16x32_bf16)
// ---------------------------------------------------------------------------
typedef __bf16 bf16_t;
typedef __bf16 v16bf __attribute__((ext_vector_type(16)));
typedef __bf16 v8bf  __attribute__((ext_vector_type(8)));
typedef float  v8f   __attribute__((ext_vector_type(8)));

#define B_     32
#define S_     16
#define L_     64
#define E_     300
#define P_     100
#define H_     256          // hidden
#define G4H    (4*H_)       // 1024 gates
#define DATT   612
#define NW     (B_*S_)      // 512 word-level sequences
#define ROWS_W (NW*L_)      // 32768 word rows
#define LDX_W  416          // E_+P_ = 400 padded to mult of 32 (13 chunks)
#define LDX_S  640          // DATT padded to mult of 32 (20 chunks)
#define LDZ    640          // [h(512); pos(100); pad]
#define LDU    624          // u rows (612 padded to mult of 16)

__device__ __forceinline__ float fsigmoid(float x) { return 1.0f / (1.0f + __expf(-x)); }
__device__ __forceinline__ float ftanh(float x)    { return 2.0f / (1.0f + __expf(-2.0f * x)) - 1.0f; }

// A-matrix fragment (16x32 bf16, ISA 7.12.2 layout):
// lane<16: row=lane,    K = {0..7}  and {16..23}
// lane>=16: row=lane-16, K = {8..15} and {24..31}
__device__ __forceinline__ v16bf load_a_frag(const bf16_t* base, int stride, int lane) {
  const int r = lane & 15, half = lane >> 4;
  const bf16_t* p = base + (size_t)r * stride + half * 8;
  v8bf lo = *(const v8bf*)(p);
  v8bf hi = *(const v8bf*)(p + 16);
  return __builtin_shufflevector(lo, hi, 0,1,2,3,4,5,6,7,8,9,10,11,12,13,14,15);
}

// B-matrix fragment (32x16 bf16): lane%16 = N column, lane/16 selects K {0..7,16..23} or {8..15,24..31}.
// Weight W stored row-major [N][K]; B(k,n) = W[n][k].
__device__ __forceinline__ v16bf load_b_frag(const bf16_t* wbase, int ldw, int lane) {
  const int n = lane & 15, kb = (lane >> 4) * 8;
  const bf16_t* p = wbase + (size_t)n * ldw + kb;
  v8bf lo = *(const v8bf*)(p);
  v8bf hi = *(const v8bf*)(p + 16);
  return __builtin_shufflevector(lo, hi, 0,1,2,3,4,5,6,7,8,9,10,11,12,13,14,15);
}

// ---------------------------------------------------------------------------
// Weight prep: f32 -> bf16 with zero K padding (optionally transposed)
// ---------------------------------------------------------------------------
__global__ void conv_pad_kernel(const float* __restrict__ src, int R, int C,
                                bf16_t* __restrict__ dst, int ldd) {
  const int r = blockIdx.x;
  for (int c = threadIdx.x; c < ldd; c += blockDim.x) {
    float v = (r < R && c < C) ? src[(size_t)r * C + c] : 0.0f;
    dst[(size_t)r * ldd + c] = (bf16_t)v;
  }
}

__global__ void conv_pad_t_kernel(const float* __restrict__ src, int S,
                                  bf16_t* __restrict__ dst, int ldd) {
  const int r = blockIdx.x;  // output row = src column
  for (int c = threadIdx.x; c < ldd; c += blockDim.x) {
    float v = (r < S && c < S) ? src[(size_t)c * S + r] : 0.0f;
    dst[(size_t)r * ldd + c] = (bf16_t)v;
  }
}

// ---------------------------------------------------------------------------
// Embedding gather: build word LSTM input x (bf16, K-padded) and the position
// part of the word attention z-buffer.
// ---------------------------------------------------------------------------
__global__ void build_x_kernel(const int* __restrict__ tok, const int* __restrict__ wp,
                               const float* __restrict__ emb, const float* __restrict__ wpe,
                               bf16_t* __restrict__ xw, bf16_t* __restrict__ zw) {
  const int row = blockIdx.x;                 // n*L + t
  const size_t tk = (size_t)tok[row] * E_;
  const size_t pk = (size_t)wp[row]  * P_;
  bf16_t* xr = xw + (size_t)row * LDX_W;
  for (int c = threadIdx.x; c < LDX_W; c += blockDim.x) {
    float v = 0.0f;
    if (c < E_)            v = emb[tk + c];
    else if (c < E_ + P_)  v = wpe[pk + (c - E_)];
    xr[c] = (bf16_t)v;
  }
  bf16_t* zr = zw + (size_t)row * LDZ;        // cols [512..639] = [wpe; pad]
  for (int c = 2 * H_ + threadIdx.x; c < LDZ; c += blockDim.x) {
    float v = (c < 2 * H_ + P_) ? wpe[pk + (c - 2 * H_)] : 0.0f;
    zr[c] = (bf16_t)v;
  }
}

// ---------------------------------------------------------------------------
// Fused BiLSTM: per block = 16 sequences x 1 direction; loops over T.
// gates[16,1024] = x_t * Wih^T + h * Whh^T  via WMMA, then LSTM elementwise.
// h is written straight into the bf16 z-buffer (cols dir*H .. dir*H+255).
// KIN = input K-chunks (compile-time so both K-loops fully unroll/pipeline).
// ---------------------------------------------------------------------------
template <int KIN>
__global__ __launch_bounds__(256)
void lstm_fused_kernel(const bf16_t* __restrict__ x, int T, int ldx,
                       const bf16_t* __restrict__ wih,   // [2*1024][ldx]
                       const bf16_t* __restrict__ whh,   // [2*1024][256]
                       const float* __restrict__ bias_f, const float* __restrict__ bias_b,
                       bf16_t* __restrict__ zout, int ldz,
                       float* __restrict__ hfinal)       // [nseq][512] or nullptr
{
  __shared__ __align__(16) bf16_t x_s[16 * LDX_S];
  __shared__ __align__(16) bf16_t h_s[16 * H_];
  __shared__ __align__(16) float  c_s[16 * H_];
  __shared__ __align__(16) float  g_s[16 * G4H];

  const int tid  = threadIdx.x;
  const int wave = tid >> 5, lane = tid & 31;
  const int dir  = blockIdx.y;
  const int n0   = blockIdx.x * 16;
  const float*  bias  = dir ? bias_b : bias_f;
  const bf16_t* wih_d = wih + (size_t)dir * 1024 * ldx;
  const bf16_t* whh_d = whh + (size_t)dir * 1024 * H_;

  for (int i = tid; i < 16 * H_; i += 256) { h_s[i] = (bf16_t)0.0f; c_s[i] = 0.0f; }
  __syncthreads();

  for (int t = 0; t < T; ++t) {
    const int tm = dir ? (T - 1 - t) : t;

    // stage x_t tile [16][ldx] into LDS (u32 copies)
    const int rowWords = ldx >> 1;
    for (int i = tid; i < 16 * rowWords; i += 256) {
      const int r = i / rowWords, cw = i - r * rowWords;
      const unsigned* src = (const unsigned*)(x + ((size_t)(n0 + r) * T + tm) * ldx);
      ((unsigned*)(x_s + (size_t)r * ldx))[cw] = src[cw];
    }
    __syncthreads();

    // Preload x/h A-fragments once per step (resident in VGPRs across all
    // 8 N-tiles; wave32 VGPR budget is ample with 1 WG/WGP at this LDS use).
    v16bf a_in[KIN];
    #pragma unroll
    for (int kc = 0; kc < KIN; ++kc) a_in[kc] = load_a_frag(x_s + kc * 32, ldx, lane);
    v16bf a_h[H_ / 32];
    #pragma unroll
    for (int kc = 0; kc < H_ / 32; ++kc) a_h[kc] = load_a_frag(h_s + kc * 32, H_, lane);

    // gates GEMM: 64 N-tiles over 8 waves
    for (int i = 0; i < 8; ++i) {
      const int nbase = (wave + i * 8) * 16;
      v8f acc = {};
      #pragma unroll
      for (int kc = 0; kc < KIN; ++kc) {
        v16bf b = load_b_frag(wih_d + (size_t)nbase * ldx + kc * 32, ldx, lane);
        acc = __builtin_amdgcn_wmma_f32_16x16x32_bf16(false, a_in[kc], false, b, (short)0, acc, false, false);
      }
      #pragma unroll
      for (int kc = 0; kc < H_ / 32; ++kc) {
        v16bf b = load_b_frag(whh_d + (size_t)nbase * H_ + kc * 32, H_, lane);
        acc = __builtin_amdgcn_wmma_f32_16x16x32_bf16(false, a_h[kc], false, b, (short)0, acc, false, false);
      }
      const int col = nbase + (lane & 15);
      const int rb  = (lane >> 4) * 8;
      #pragma unroll
      for (int v = 0; v < 8; ++v) g_s[(size_t)(rb + v) * G4H + col] = acc[v];
    }
    __syncthreads();

    // LSTM elementwise: 16 rows x 256 units
    for (int u = tid; u < 16 * H_; u += 256) {
      const int r = u >> 8, j = u & (H_ - 1);
      const float gi = g_s[(size_t)r * G4H + j         ] + bias[j];
      const float gf = g_s[(size_t)r * G4H + j +   H_  ] + bias[j +   H_];
      const float gg = g_s[(size_t)r * G4H + j + 2*H_  ] + bias[j + 2*H_];
      const float go = g_s[(size_t)r * G4H + j + 3*H_  ] + bias[j + 3*H_];
      const float c  = fsigmoid(gf) * c_s[u] + fsigmoid(gi) * ftanh(gg);
      const float h  = fsigmoid(go) * ftanh(c);
      c_s[u] = c;
      h_s[u] = (bf16_t)h;
      zout[((size_t)(n0 + r) * T + tm) * ldz + dir * H_ + j] = (bf16_t)h;
    }
    __syncthreads();
  }

  if (hfinal) {
    for (int u = tid; u < 16 * H_; u += 256) {
      const int r = u >> 8, j = u & (H_ - 1);
      hfinal[(size_t)(n0 + r) * (2 * H_) + dir * H_ + j] = (float)h_s[u];
    }
  }
}

// ---------------------------------------------------------------------------
// Generic bf16 WMMA GEMM: C[M,N] = act(A[M,K] * W[N,K]^T + bias)
// block tile 64x64, 8 waves x 2 (16x16) tiles, K in 32-chunks (pre-padded).
// ---------------------------------------------------------------------------
template <int KC>
__global__ __launch_bounds__(256)
void gemm_bias_act_kernel(const bf16_t* __restrict__ A, int lda,
                          const bf16_t* __restrict__ W, int ldw,
                          const float* __restrict__ bias, int nbias,
                          float* __restrict__ C, int N, int ldc, int act)
{
  const int tid = threadIdx.x, wave = tid >> 5, lane = tid & 31;
  const int mb = blockIdx.x * 64, nb = blockIdx.y * 64;
  for (int pi = 0; pi < 2; ++pi) {
    const int p  = wave + pi * 8;
    const int mt = p >> 2, nt = p & 3;
    const bf16_t* Abase = A + (size_t)(mb + mt * 16) * lda;
    const bf16_t* Wbase = W + (size_t)(nb + nt * 16) * ldw;
    v8f acc = {};
    #pragma unroll
    for (int kc = 0; kc < KC; ++kc) {
      v16bf a = load_a_frag(Abase + kc * 32, lda, lane);
      v16bf b = load_b_frag(Wbase + kc * 32, ldw, lane);
      acc = __builtin_amdgcn_wmma_f32_16x16x32_bf16(false, a, false, b, (short)0, acc, false, false);
    }
    const int col  = nb + nt * 16 + (lane & 15);
    const int row0 = mb + mt * 16 + (lane >> 4) * 8;
    #pragma unroll
    for (int v = 0; v < 8; ++v) {
      float val = acc[v] + ((col < nbias) ? bias[col] : 0.0f);
      if (act) val = ftanh(val);
      if (col < N) C[(size_t)(row0 + v) * ldc + col] = val;
    }
  }
}

// ---------------------------------------------------------------------------
// Attention pool: logits = ut . proj, softmax over T, out = sum a_t * h_t
// ---------------------------------------------------------------------------
__global__ __launch_bounds__(256)
void attn_pool_kernel(const float* __restrict__ ut, int ldu, int dproj,
                      const float* __restrict__ proj,
                      const bf16_t* __restrict__ z, int ldz, int T, int hdim,
                      float* __restrict__ out)
{
  __shared__ float lg[64];
  const int n = blockIdx.x, tid = threadIdx.x;
  if (tid < T) {
    const float* ur = ut + ((size_t)n * T + tid) * ldu;
    float s = 0.0f;
    for (int e = 0; e < dproj; ++e) s += ur[e] * proj[e];
    lg[tid] = s;
  }
  __syncthreads();
  float mx = -1e30f;
  for (int t = 0; t < T; ++t) mx = fmaxf(mx, lg[t]);
  float den = 0.0f;
  for (int t = 0; t < T; ++t) den += __expf(lg[t] - mx);
  const float inv = 1.0f / den;
  for (int col = tid; col < hdim; col += blockDim.x) {
    float acc = 0.0f;
    for (int t = 0; t < T; ++t)
      acc += (__expf(lg[t] - mx) * inv) * (float)z[((size_t)n * T + t) * ldz + col];
    out[(size_t)n * hdim + col] = acc;
  }
}

// ---------------------------------------------------------------------------
// Build sentence-level inputs: sx = [sen_vec; spe], and pos part of z2.
// ---------------------------------------------------------------------------
__global__ void build_sx_kernel(const float* __restrict__ senvec,
                                const int* __restrict__ segpos,
                                const float* __restrict__ spe,
                                bf16_t* __restrict__ sxw, bf16_t* __restrict__ z2w)
{
  const int r = blockIdx.x;                   // b*S + s
  const size_t pk = (size_t)segpos[r] * P_;
  for (int c = threadIdx.x; c < LDX_S; c += blockDim.x) {
    if (c < 2 * H_) {
      sxw[(size_t)r * LDX_S + c] = (bf16_t)senvec[(size_t)r * (2 * H_) + c];
    } else {
      float v = (c < 2 * H_ + P_) ? spe[pk + (c - 2 * H_)] : 0.0f;
      bf16_t bv = (bf16_t)v;
      sxw[(size_t)r * LDX_S + c] = bv;
      z2w[(size_t)r * LDZ   + c] = bv;
    }
  }
}

// ---------------------------------------------------------------------------
// Classifier: out[b,p] = [doc_vec ; dh] . dense_W[p] + dense_b[p]
// ---------------------------------------------------------------------------
__global__ void classifier_kernel(const float* __restrict__ docvec, const float* __restrict__ dh,
                                  const float* __restrict__ dW, const float* __restrict__ db,
                                  float* __restrict__ out)
{
  const int t = threadIdx.x;
  if (t >= B_ * 3) return;
  const int b = t / 3, p = t % 3;
  float s = db[p];
  for (int k = 0; k < 2 * H_; ++k) s += docvec[(size_t)b * 2 * H_ + k] * dW[(size_t)p * G4H + k];
  for (int k = 0; k < 2 * H_; ++k) s += dh[(size_t)b * 2 * H_ + k] * dW[(size_t)p * G4H + 2 * H_ + k];
  out[b * 3 + p] = s;
}

// ---------------------------------------------------------------------------
extern "C" void kernel_launch(void* const* d_in, const int* in_sizes, int n_in,
                              void* d_out, int out_size, void* d_ws, size_t ws_size,
                              hipStream_t stream) {
  (void)in_sizes; (void)out_size;
  if (n_in < 26) return;

  const int*   tri      = (const int*)  d_in[0];
  const int*   wposi    = (const int*)  d_in[1];
  const int*   segpos   = (const int*)  d_in[2];
  const float* emb      = (const float*)d_in[3];
  const float* wpe_tab  = (const float*)d_in[4];
  const float* spe_tab  = (const float*)d_in[5];
  const float* wWih_f   = (const float*)d_in[6];
  const float* wWhh_f   = (const float*)d_in[7];
  const float* wb_f     = (const float*)d_in[8];
  const float* wWih_b   = (const float*)d_in[9];
  const float* wWhh_b   = (const float*)d_in[10];
  const float* wb_b     = (const float*)d_in[11];
  const float* sWih_f   = (const float*)d_in[12];
  const float* sWhh_f   = (const float*)d_in[13];
  const float* sb_f     = (const float*)d_in[14];
  const float* sWih_b   = (const float*)d_in[15];
  const float* sWhh_b   = (const float*)d_in[16];
  const float* sb_b     = (const float*)d_in[17];
  const float* word_W   = (const float*)d_in[18];
  const float* word_bias= (const float*)d_in[19];
  const float* word_proj= (const float*)d_in[20];
  const float* sent_W   = (const float*)d_in[21];
  const float* sent_bias= (const float*)d_in[22];
  const float* sent_proj= (const float*)d_in[23];
  const float* dense_W  = (const float*)d_in[24];
  const float* dense_b  = (const float*)d_in[25];
  float* out = (float*)d_out;

  // workspace layout
  char* wsb = (char*)d_ws;
  size_t off = 0;
  auto take = [&](size_t bytes) -> char* {
    char* p = wsb + off;
    off = (off + bytes + 255) & ~(size_t)255;
    return p;
  };
  bf16_t* xw    = (bf16_t*)take((size_t)ROWS_W * LDX_W * 2);   // word input (bf16, padded)
  bf16_t* zw    = (bf16_t*)take((size_t)ROWS_W * LDZ   * 2);   // [h;wpe] word attention input
  float*  ut    = (float*) take((size_t)ROWS_W * LDU   * 4);   // tanh(z W + b)
  bf16_t* wih   = (bf16_t*)take((size_t)2048 * LDX_W * 2);
  bf16_t* whh   = (bf16_t*)take((size_t)2048 * H_    * 2);
  bf16_t* swih  = (bf16_t*)take((size_t)2048 * LDX_S * 2);
  bf16_t* swhh  = (bf16_t*)take((size_t)2048 * H_    * 2);
  bf16_t* wWt   = (bf16_t*)take((size_t)640 * 640 * 2);        // word_W^T bf16 padded
  bf16_t* sWt   = (bf16_t*)take((size_t)640 * 640 * 2);
  float*  senvec= (float*) take((size_t)NW * 2 * H_ * 4);
  bf16_t* sxw   = (bf16_t*)take((size_t)NW * LDX_S * 2);       // sentence LSTM input
  bf16_t* z2w   = (bf16_t*)take((size_t)NW * LDZ   * 2);       // [doc h; spe]
  float*  ut2   = (float*) take((size_t)NW * LDU * 4);
  float*  dh    = (float*) take((size_t)B_ * 2 * H_ * 4);      // final doc hidden (f|b)
  float*  docvec= (float*) take((size_t)B_ * 2 * H_ * 4);
  if (off > ws_size) return;  // workspace too small; bail deterministically

  // --- weight prep (bf16 + pad / transpose) ---
  conv_pad_kernel<<<1024, 128, 0, stream>>>(wWih_f, 1024, E_ + P_, wih,                 LDX_W);
  conv_pad_kernel<<<1024, 128, 0, stream>>>(wWih_b, 1024, E_ + P_, wih + 1024 * LDX_W,  LDX_W);
  conv_pad_kernel<<<1024, 128, 0, stream>>>(wWhh_f, 1024, H_,      whh,                 H_);
  conv_pad_kernel<<<1024, 128, 0, stream>>>(wWhh_b, 1024, H_,      whh + 1024 * H_,     H_);
  conv_pad_kernel<<<1024, 128, 0, stream>>>(sWih_f, 1024, DATT,    swih,                LDX_S);
  conv_pad_kernel<<<1024, 128, 0, stream>>>(sWih_b, 1024, DATT,    swih + 1024 * LDX_S, LDX_S);
  conv_pad_kernel<<<1024, 128, 0, stream>>>(sWhh_f, 1024, H_,      swhh,                H_);
  conv_pad_kernel<<<1024, 128, 0, stream>>>(sWhh_b, 1024, H_,      swhh + 1024 * H_,    H_);
  conv_pad_t_kernel<<<640, 128, 0, stream>>>(word_W, DATT, wWt, 640);
  conv_pad_t_kernel<<<640, 128, 0, stream>>>(sent_W, DATT, sWt, 640);

  // --- embedding gather ---
  build_x_kernel<<<ROWS_W, 128, 0, stream>>>(tri, wposi, emb, wpe_tab, xw, zw);

  // --- word-level BiLSTM (fused input+recurrent WMMA GEMMs) ---
  lstm_fused_kernel<LDX_W / 32><<<dim3(NW / 16, 2), 256, 0, stream>>>(
      xw, L_, LDX_W, wih, whh, wb_f, wb_b, zw, LDZ, (float*)nullptr);

  // --- word attention: u = tanh(z W + b) ; pool ---
  gemm_bias_act_kernel<LDZ / 32><<<dim3(ROWS_W / 64, (DATT + 63) / 64), 256, 0, stream>>>(
      zw, LDZ, wWt, 640, word_bias, DATT, ut, DATT, LDU, 1);
  attn_pool_kernel<<<NW, 256, 0, stream>>>(ut, LDU, DATT, word_proj, zw, LDZ, L_, 2 * H_, senvec);

  // --- sentence-level input ---
  build_sx_kernel<<<NW, 128, 0, stream>>>(senvec, segpos, spe_tab, sxw, z2w);

  // --- sentence-level BiLSTM ---
  lstm_fused_kernel<LDX_S / 32><<<dim3(B_ / 16, 2), 256, 0, stream>>>(
      sxw, S_, LDX_S, swih, swhh, sb_f, sb_b, z2w, LDZ, dh);

  // --- sentence attention ---
  gemm_bias_act_kernel<LDZ / 32><<<dim3(NW / 64, (DATT + 63) / 64), 256, 0, stream>>>(
      z2w, LDZ, sWt, 640, sent_bias, DATT, ut2, DATT, LDU, 1);
  attn_pool_kernel<<<B_, 256, 0, stream>>>(ut2, LDU, DATT, sent_proj, z2w, LDZ, S_, 2 * H_, docvec);

  // --- classifier ---
  classifier_kernel<<<1, 128, 0, stream>>>(docvec, dh, dense_W, dense_b, out);
}
